// HAN_57758720197062
// MI455X (gfx1250) — matrix operand order repaired and checked by reading, not compiled, not run on hardware
//
#include <hip/hip_runtime.h>
#include <math.h>

#define NNODES 50000
#define NEDGES 500000
#define FEAT   128
#define NHEAD  8
#define DHEAD  16
#define NEG_SLOPE 0.2f

typedef __attribute__((ext_vector_type(16))) _Float16 v16h;
typedef __attribute__((ext_vector_type(8)))  float    v8f;

static __host__ __device__ inline int cdiv(int a, int b) { return (a + b - 1) / b; }

// ---- order-preserving float <-> uint map for atomicMax-based segment_max ----
static __device__ __forceinline__ unsigned f32_order(float f) {
    unsigned u = __float_as_uint(f);
    return (u & 0x80000000u) ? ~u : (u | 0x80000000u);
}
static __device__ __forceinline__ float f32_unorder(unsigned o) {
    unsigned u = (o & 0x80000000u) ? (o & 0x7fffffffu) : ~o;
    return __uint_as_float(u);
}

__global__ void zero_f32(float* __restrict__ p, int n) {
    int i = blockIdx.x * blockDim.x + threadIdx.x;
    if (i < n) p[i] = 0.0f;
}

// ---- convert W[k][n] f32 row-major -> Wt[n][k] f16 (column-major of W) ----
// Makes each WMMA B-fragment a single contiguous 32-byte load.
__global__ void conv_w_f16t(const float* __restrict__ W, _Float16* __restrict__ Wt) {
    int t = blockIdx.x * blockDim.x + threadIdx.x;
    if (t >= FEAT * FEAT) return;
    int n = t >> 7, k = t & (FEAT - 1);
    Wt[t] = (_Float16)W[k * FEAT + n];
}

// ---------------------------------------------------------------------------
// WMMA GEMM: D[M,128] = A[M,128] @ W[128,128] + bias      (EPI == 0)
// EPI == 1: no store; accumulate  sum_rows tanh(acc+b) . q  into *score
// One wave computes one 16x16 tile; K=128 as 4x v_wmma_f32_16x16x32_f16.
// Fragment layouts used here (wave32, contiguous k order in A and B):
//   A 16x32 f16 : lane&15 = M row; half=lane>>4; a[j] = A[m][k0+16*half+j]
//   B 32x16 f16 : lane&15 = N col; b[j] = W[k0+16*half+j][n] = Wt[n][k0+16*half+j]
//   C/D 16x16 f32: lane&15 = N col; acc[r] = row m0 + 8*half + r
// ---------------------------------------------------------------------------
template<int EPI>
__global__ void gemm128_wmma(const float* __restrict__ A,
                             const _Float16* __restrict__ Wt,
                             const float* __restrict__ bias,
                             float* __restrict__ D,
                             const float* __restrict__ q,
                             float* __restrict__ score,
                             int M)
{
    const int gtid   = blockIdx.x * blockDim.x + threadIdx.x;
    const int wave   = gtid >> 5;
    const int lane   = threadIdx.x & 31;
    const int tile_m = wave >> 3;          // 8 N-tiles across 128 columns
    const int tile_n = wave & 7;
    if (tile_m * 16 >= M) return;
    const int m0 = tile_m * 16, n0 = tile_n * 16;
    const int l15 = lane & 15, half = lane >> 4;
    const int n = n0 + l15;

    v8f acc = {};
    const float*    arow = A  + (size_t)(m0 + l15) * FEAT;
    const _Float16* wcol = Wt + (size_t)n * FEAT;

    #pragma unroll
    for (int k0 = 0; k0 < FEAT; k0 += 32) {
        const float4 f0 = *(const float4*)(arow + k0 + half * 16);
        const float4 f1 = *(const float4*)(arow + k0 + half * 16 + 4);
        const float4 f2 = *(const float4*)(arow + k0 + half * 16 + 8);
        const float4 f3 = *(const float4*)(arow + k0 + half * 16 + 12);
        v16h a;
        a[0]=(_Float16)f0.x;  a[1]=(_Float16)f0.y;  a[2]=(_Float16)f0.z;  a[3]=(_Float16)f0.w;
        a[4]=(_Float16)f1.x;  a[5]=(_Float16)f1.y;  a[6]=(_Float16)f1.z;  a[7]=(_Float16)f1.w;
        a[8]=(_Float16)f2.x;  a[9]=(_Float16)f2.y;  a[10]=(_Float16)f2.z; a[11]=(_Float16)f2.w;
        a[12]=(_Float16)f3.x; a[13]=(_Float16)f3.y; a[14]=(_Float16)f3.z; a[15]=(_Float16)f3.w;

        const v16h b = *(const v16h*)(wcol + k0 + half * 16);

        acc = __builtin_amdgcn_wmma_f32_16x16x32_f16(false, a, false, b,
                                                     (short)0, acc, false, false);
    }

    const float bn = bias[n];
    if (EPI == 0) {
        #pragma unroll
        for (int r = 0; r < 8; ++r) {
            const int m = m0 + (half << 3) + r;
            D[(size_t)m * FEAT + n] = acc[r] + bn;
        }
    } else {
        const float qn = q[n];
        float s = 0.0f;
        #pragma unroll
        for (int r = 0; r < 8; ++r) s += tanhf(acc[r] + bn) * qn;
        #pragma unroll
        for (int off = 16; off > 0; off >>= 1) s += __shfl_xor(s, off, 32);
        if (lane == 0) atomicAdd(score, s);
    }
}

// ---- per-node attention coefficients:  out[n,h] = sum_d h[n,h,d]*att[h,d] ----
__global__ void node_coeff(const float* __restrict__ h, const float* __restrict__ att,
                           float* __restrict__ out) {
    int t = blockIdx.x * blockDim.x + threadIdx.x;
    if (t >= NNODES * NHEAD) return;
    int n = t >> 3, hh = t & 7;
    const float* hp = h + (size_t)n * FEAT + hh * DHEAD;
    const float* ap = att + hh * DHEAD;
    float s = 0.0f;
    #pragma unroll
    for (int d = 0; d < DHEAD; ++d) s += hp[d] * ap[d];
    out[t] = s;
}

// ---- pass 1: alpha = leaky_relu(as[src]+ad[dst]); segment max via atomicMax ----
__global__ void edge_alpha_max(const int* __restrict__ ei,
                               const float* __restrict__ asn,
                               const float* __restrict__ adn,
                               float* __restrict__ alpha,
                               unsigned* __restrict__ amax) {
    int e = blockIdx.x * blockDim.x + threadIdx.x;
    if (e >= NEDGES) return;
    int s = ei[e], d = ei[NEDGES + e];
    #pragma unroll
    for (int hh = 0; hh < NHEAD; ++hh) {
        float a = asn[s * NHEAD + hh] + adn[d * NHEAD + hh];
        a = a > 0.0f ? a : NEG_SLOPE * a;
        alpha[(size_t)e * NHEAD + hh] = a;
        atomicMax(&amax[d * NHEAD + hh], f32_order(a));
    }
}

// ---- pass 2: ex = exp(alpha - amax[dst]); denom += ex; out[dst] += h_src*ex ----
__global__ void edge_scatter(const int* __restrict__ ei,
                             const float* __restrict__ alpha,
                             const unsigned* __restrict__ amax,
                             const float* __restrict__ hsrc,
                             float* __restrict__ denom,
                             float* __restrict__ out) {
    int t = blockIdx.x * blockDim.x + threadIdx.x;
    if (t >= NEDGES * NHEAD) return;
    int e = t >> 3, hh = t & 7;
    int s = ei[e], d = ei[NEDGES + e];
    float am = f32_unorder(amax[d * NHEAD + hh]);
    float ex = expf(alpha[t] - am);
    atomicAdd(&denom[d * NHEAD + hh], ex);
    const float* hp = hsrc + (size_t)s * FEAT + hh * DHEAD;
    float*       op = out  + (size_t)d * FEAT + hh * DHEAD;
    #pragma unroll
    for (int k = 0; k < DHEAD; ++k) atomicAdd(&op[k], hp[k] * ex);
}

// ---- pass 3: normalize by segment sum, relu ----
__global__ void node_norm_relu(const float* __restrict__ denom, float* __restrict__ out) {
    int t = blockIdx.x * blockDim.x + threadIdx.x;
    if (t >= NNODES * FEAT) return;
    int n = t >> 7, hh = (t >> 4) & 7;
    float v = out[t] / (denom[n * NHEAD + hh] + 1e-16f);
    out[t] = v > 0.0f ? v : 0.0f;
}

// ---- semantic combine: softmax over 2 metapath scores, weighted sum, opt ELU ----
__global__ void combine_sem(const float* __restrict__ o0, const float* __restrict__ o1,
                            const float* __restrict__ scores, float* __restrict__ dst,
                            int do_elu) {
    int t = blockIdx.x * blockDim.x + threadIdx.x;
    if (t >= NNODES * FEAT) return;
    float s0 = scores[0] * (1.0f / NNODES);
    float s1 = scores[1] * (1.0f / NNODES);
    float m  = fmaxf(s0, s1);
    float e0 = expf(s0 - m), e1 = expf(s1 - m);
    float inv = 1.0f / (e0 + e1);
    float v = (e0 * inv) * o0[t] + (e1 * inv) * o1[t];
    if (do_elu) v = v > 0.0f ? v : expm1f(v);
    dst[t] = v;
}

// ---------------------------------------------------------------------------
static void run_layer(const float* xc, const float* xd,
                      const int* const* eis,
                      const float* pW, const float* pb,
                      const float* att_s, const float* att_d,
                      const float* kW, const float* kb, const float* q,
                      float* hC, float* hD, float* const* O,
                      float* asn, float* adn, float* alpha,
                      unsigned* amaxU, float* denom, float* scores,
                      _Float16* wt0, _Float16* wt1, _Float16* wtK,
                      float* destC, float* destD, int do_elu,
                      hipStream_t st)
{
    const int BLK = 256;
    const int gemm_blocks = cdiv((NNODES / 16) * 8 * 32, BLK);   // 3125
    const int wblk = cdiv(FEAT * FEAT, BLK);

    // pre-transpose+convert weights to f16 column-major (one B-fragment = 1 load)
    conv_w_f16t<<<wblk, BLK, 0, st>>>(pW,               wt0);
    conv_w_f16t<<<wblk, BLK, 0, st>>>(pW + FEAT * FEAT, wt1);
    conv_w_f16t<<<wblk, BLK, 0, st>>>(kW,               wtK);

    // projections: h = x @ W + b  (per node type)
    gemm128_wmma<0><<<gemm_blocks, BLK, 0, st>>>(xc, wt0, pb,        hC, nullptr, nullptr, NNODES);
    gemm128_wmma<0><<<gemm_blocks, BLK, 0, st>>>(xd, wt1, pb + FEAT, hD, nullptr, nullptr, NNODES);

    // edge types: 0 c->d, 1 d->c, 2 c->c, 3 d->d
    const float* srcH[4] = { hC, hD, hC, hD };
    const float* dstH[4] = { hD, hC, hC, hD };
    for (int t = 0; t < 4; ++t) {
        node_coeff<<<cdiv(NNODES * NHEAD, BLK), BLK, 0, st>>>(srcH[t], att_s + t * FEAT, asn);
        node_coeff<<<cdiv(NNODES * NHEAD, BLK), BLK, 0, st>>>(dstH[t], att_d + t * FEAT, adn);
        zero_f32<<<cdiv(NNODES * NHEAD, BLK), BLK, 0, st>>>((float*)amaxU, NNODES * NHEAD);
        zero_f32<<<cdiv(NNODES * NHEAD, BLK), BLK, 0, st>>>(denom, NNODES * NHEAD);
        zero_f32<<<cdiv(NNODES * FEAT, BLK), BLK, 0, st>>>(O[t], NNODES * FEAT);
        edge_alpha_max<<<cdiv(NEDGES, BLK), BLK, 0, st>>>(eis[t], asn, adn, alpha, amaxU);
        edge_scatter<<<cdiv(NEDGES * NHEAD, BLK), BLK, 0, st>>>(eis[t], alpha, amaxU, srcH[t], denom, O[t]);
        node_norm_relu<<<cdiv(NNODES * FEAT, BLK), BLK, 0, st>>>(denom, O[t]);
    }

    // semantic-level attention
    zero_f32<<<1, 4, 0, st>>>(scores, 4);
    gemm128_wmma<1><<<gemm_blocks, BLK, 0, st>>>(O[1], wtK, kb, nullptr, q, scores + 0, NNODES); // dc (circ)
    gemm128_wmma<1><<<gemm_blocks, BLK, 0, st>>>(O[2], wtK, kb, nullptr, q, scores + 1, NNODES); // cc (circ)
    gemm128_wmma<1><<<gemm_blocks, BLK, 0, st>>>(O[0], wtK, kb, nullptr, q, scores + 2, NNODES); // cd (dis)
    gemm128_wmma<1><<<gemm_blocks, BLK, 0, st>>>(O[3], wtK, kb, nullptr, q, scores + 3, NNODES); // dd (dis)
    combine_sem<<<cdiv(NNODES * FEAT, BLK), BLK, 0, st>>>(O[1], O[2], scores + 0, destC, do_elu);
    combine_sem<<<cdiv(NNODES * FEAT, BLK), BLK, 0, st>>>(O[0], O[3], scores + 2, destD, do_elu);
}

extern "C" void kernel_launch(void* const* d_in, const int* in_sizes, int n_in,
                              void* d_out, int out_size, void* d_ws, size_t ws_size,
                              hipStream_t stream)
{
    (void)in_sizes; (void)n_in; (void)out_size; (void)ws_size;

    const float* x_circ   = (const float*)d_in[0];
    const float* x_dis    = (const float*)d_in[1];
    const int*   eis[4]   = { (const int*)d_in[2], (const int*)d_in[3],
                              (const int*)d_in[4], (const int*)d_in[5] };
    const float* p1_W     = (const float*)d_in[6];
    const float* p1_b     = (const float*)d_in[7];
    const float* att1_src = (const float*)d_in[8];
    const float* att1_dst = (const float*)d_in[9];
    const float* k1_W     = (const float*)d_in[10];
    const float* k1_b     = (const float*)d_in[11];
    const float* q1       = (const float*)d_in[12];
    const float* p2_W     = (const float*)d_in[13];
    const float* p2_b     = (const float*)d_in[14];
    const float* att2_src = (const float*)d_in[15];
    const float* att2_dst = (const float*)d_in[16];
    const float* k2_W     = (const float*)d_in[17];
    const float* k2_b     = (const float*)d_in[18];
    const float* q2       = (const float*)d_in[19];

    const size_t NF = (size_t)NNODES * FEAT;
    float* ws = (float*)d_ws;
    float* hC = ws;            float* hD = hC + NF;
    float* O0 = hD + NF;       float* O1 = O0 + NF;
    float* O2 = O1 + NF;       float* O3 = O2 + NF;
    float* C0 = O3 + NF;       float* C1 = C0 + NF;
    float* Obuf[4] = { O0, O1, O2, O3 };
    float* asn   = C1 + NF;
    float* adn   = asn + (size_t)NNODES * NHEAD;
    float* alpha = adn + (size_t)NNODES * NHEAD;
    unsigned* amaxU = (unsigned*)(alpha + (size_t)NEDGES * NHEAD);
    float* denom  = (float*)amaxU + (size_t)NNODES * NHEAD;
    float* scores = denom + (size_t)NNODES * NHEAD;
    _Float16* wt0 = (_Float16*)(scores + 16);
    _Float16* wt1 = wt0 + FEAT * FEAT;
    _Float16* wtK = wt1 + FEAT * FEAT;

    float* out_c = (float*)d_out;
    float* out_d = out_c + NF;

    // layer 1 (ELU on outputs), results into C0/C1
    run_layer(x_circ, x_dis, eis, p1_W, p1_b, att1_src, att1_dst, k1_W, k1_b, q1,
              hC, hD, Obuf, asn, adn, alpha, amaxU, denom, scores,
              wt0, wt1, wtK, C0, C1, /*do_elu=*/1, stream);

    // layer 2 (no ELU), results straight into d_out [2,N,128]
    run_layer(C0, C1, eis, p2_W, p2_b, att2_src, att2_dst, k2_W, k2_b, q2,
              hC, hD, Obuf, asn, adn, alpha, amaxU, denom, scores,
              wt0, wt1, wtK, out_c, out_d, /*do_elu=*/0, stream);
}
